// LinearAttention_74500502716607
// MI455X (gfx1250) — matrix operand order repaired and checked by reading, not compiled, not run on hardware
//
#include <hip/hip_runtime.h>
#include <cstdint>

// ---------------------------------------------------------------------------
// Types for CDNA5 WMMA (wave32): v_wmma_f32_16x16x32_bf16
// ---------------------------------------------------------------------------
typedef __attribute__((ext_vector_type(16))) __bf16 v16bf;
typedef __attribute__((ext_vector_type(8)))  __bf16 v8bf;
typedef __attribute__((ext_vector_type(8)))  float  v8f;

__device__ __forceinline__ v16bf cat8(v8bf lo, v8bf hi) {
  return __builtin_shufflevector(lo, hi, 0,1,2,3,4,5,6,7,8,9,10,11,12,13,14,15);
}

// D = A(16x32 bf16) * B(32x16 bf16) + C(16x16 f32)
__device__ __forceinline__ v8f wmma_bf16(v16bf a, v16bf b, v8f c) {
  return __builtin_amdgcn_wmma_f32_16x16x32_bf16(
      /*neg_a=*/false, a, /*neg_b=*/false, b,
      /*c_mod=*/(short)0, c, /*reuse_a=*/false, /*reuse_b=*/false);
}

// A fragment: matrix stored row-major [m][k] in LDS (bf16).
// k order used here, not the ISA table order (B matches): lane m = lane&15, half = lane>>4,
//   elems 0..15 = K[16*half .. 16*half+15]  (contiguous in k)
__device__ __forceinline__ v16bf ldsA(const __bf16* lds, int stride,
                                      int row, int kbase, int lane) {
  const __bf16* p = lds + row * stride + kbase;
  const int half = lane >> 4;
  v8bf lo = *(const v8bf*)(p + 16 * half);
  v8bf hi = *(const v8bf*)(p + 16 * half + 8);
  return cat8(lo, hi);
}

// B fragment: operand matrix B is (K x N); we stage it TRANSPOSED in LDS as
// BT[n][k] (row-major, k contiguous). Lane n = lane&15, half = lane>>4,
//   elems 0..15 = K[16*half .. 16*half+15]  (contiguous in k)
__device__ __forceinline__ v16bf ldsB(const __bf16* lds, int stride,
                                      int col, int kbase, int lane) {
  const __bf16* p = lds + col * stride + kbase + 16 * (lane >> 4);
  v8bf lo = *(const v8bf*)(p);
  v8bf hi = *(const v8bf*)(p + 8);
  return cat8(lo, hi);
}

// Same as ldsB but source is an fp32 LDS tile (KV state): convert on read.
__device__ __forceinline__ v16bf ldsB_f32(const float* lds, int stride,
                                          int col, int kbase, int lane) {
  const float* p = lds + col * stride + kbase + 16 * (lane >> 4);
  v16bf r;
#pragma unroll
  for (int i = 0; i < 16; ++i) r[i] = (__bf16)p[i];
  return r;
}

// ---------------------------------------------------------------------------
// Problem constants
// ---------------------------------------------------------------------------
#define BATCH 2
#define SEQ   2048
#define DMODEL 1024
#define NHEAD 16
#define HDIM  64
#define CHUNK 64
#define EPSV  1e-6f

// ---------------------------------------------------------------------------
// GEMM: Y[m,n] = sum_k X[m,k] * W[n,k] + bias[n]
//   MODE 0: write fp32 out[m*N+n]                     (final output proj)
//   MODE 1: write bf16 [B,H,S,hd] with phi=elu(x)+1   (Q, K)
//   MODE 2: write bf16 [B,H,S,hd] plain               (V)
// Tile: 128(M) x 64(N), BK=32, 8 waves of 32 (each 32x32 via 2x2 WMMA tiles)
// ---------------------------------------------------------------------------
template <int MODE>
__global__ __launch_bounds__(256) void gemm_bf16_kernel(
    const float* __restrict__ X, const float* __restrict__ W,
    const float* __restrict__ bias, void* __restrict__ outv,
    int M, int N, int K) {
  constexpr int XSTR = 40;  // 32 + 8 pad (bf16 elems) -> 80B rows, 16B aligned
  __shared__ __bf16 Xs[128 * XSTR];
  __shared__ __bf16 Ws[64 * XSTR];

  const int m0 = blockIdx.y * 128;
  const int n0 = blockIdx.x * 64;
  const int tid = threadIdx.x;
  const int lane = tid & 31;
  const int wid = tid >> 5;
  const int wm = (wid & 3) * 32;  // 4 waves along M
  const int wn = (wid >> 2) * 32; // 2 waves along N

  v8f acc[2][2] = {};

  for (int kk = 0; kk < K; kk += 32) {
    // stage X tile 128x32 (fp32 -> bf16), coalesced along k
#pragma unroll
    for (int e = 0; e < 16; ++e) {
      int idx = tid + e * 256;
      int r = idx >> 5, c = idx & 31;
      Xs[r * XSTR + c] = (__bf16)X[(size_t)(m0 + r) * K + kk + c];
    }
    // stage W tile 64x32 (already [n][k] = BT layout)
#pragma unroll
    for (int e = 0; e < 8; ++e) {
      int idx = tid + e * 256;
      int r = idx >> 5, c = idx & 31;
      Ws[r * XSTR + c] = (__bf16)W[(size_t)(n0 + r) * K + kk + c];
    }
    __syncthreads();
#pragma unroll
    for (int i = 0; i < 2; ++i) {
      v16bf a = ldsA(Xs, XSTR, wm + i * 16 + (lane & 15), 0, lane);
#pragma unroll
      for (int j = 0; j < 2; ++j) {
        v16bf b = ldsB(Ws, XSTR, wn + j * 16 + (lane & 15), 0, lane);
        acc[i][j] = wmma_bf16(a, b, acc[i][j]);
      }
    }
    __syncthreads();
  }

  // epilogue: C/D layout -> m = tile + r + 8*(lane>>4), n = tile + (lane&15)
#pragma unroll
  for (int i = 0; i < 2; ++i) {
#pragma unroll
    for (int j = 0; j < 2; ++j) {
      const int nn = n0 + wn + j * 16 + (lane & 15);
#pragma unroll
      for (int r = 0; r < 8; ++r) {
        const int mm = m0 + wm + i * 16 + r + 8 * (lane >> 4);
        float v = acc[i][j][r] + bias[nn];
        if (MODE == 0) {
          ((float*)outv)[(size_t)mm * N + nn] = v;
        } else {
          if (MODE == 1) v = (v > 0.f) ? (v + 1.f) : __expf(v);  // elu(x)+1
          const int b = mm >> 11;        // / SEQ
          const int s = mm & (SEQ - 1);
          const int h = nn >> 6;         // / HDIM
          const int d = nn & (HDIM - 1);
          ((__bf16*)outv)[(((size_t)(b * NHEAD + h)) * SEQ + s) * HDIM + d] =
              (__bf16)v;
        }
      }
    }
  }
}

// ---------------------------------------------------------------------------
// Chunked causal linear attention. One workgroup per (b,h); 32 chunks of 64.
// Per chunk (Q,K,V are 64x64):
//   S  = tril(Q K^T)                       (WMMA, masked, -> bf16 LDS)
//   den[t] = q_t . k_cum + rowsum(S[t,:])  (fp32)
//   num = Q @ KV + S @ V                   (WMMA, fp32 acc)
//   KV += K^T V ; k_cum += colsum(K)       (WMMA + fp32 state)
//   O[t,:] = num[t,:] / (den[t] + eps)
// ---------------------------------------------------------------------------
__global__ __launch_bounds__(256) void linattn_kernel(
    const __bf16* __restrict__ Q, const __bf16* __restrict__ K,
    const __bf16* __restrict__ V, float* __restrict__ O) {
  constexpr int STR = 72;   // bf16 row stride (144B, 16B aligned)
  constexpr int KVS = 68;   // fp32 row stride
  __shared__ __bf16 Qs[CHUNK * STR];   // [t][d]
  __shared__ __bf16 Ks[CHUNK * STR];   // [s][d]
  __shared__ __bf16 Kts[HDIM * STR];   // [d][s]   (A for K^T V, rows for kcum)
  __shared__ __bf16 Vts[HDIM * STR];   // [e][s]   (BT layout of V)
  __shared__ __bf16 Ss[CHUNK * STR];   // [t][s]   masked scores, bf16
  __shared__ float  KVt[HDIM * KVS];   // [e][d]   (BT layout of KV state)
  __shared__ float  kcum[HDIM];
  __shared__ float  den[CHUNK];

  const int bh = blockIdx.x;          // b*NHEAD + h
  const int b = bh >> 4;
  const int h = bh & 15;
  const __bf16* Qg = Q + (size_t)bh * SEQ * HDIM;
  const __bf16* Kg = K + (size_t)bh * SEQ * HDIM;
  const __bf16* Vg = V + (size_t)bh * SEQ * HDIM;
  float* Og = O + (size_t)b * SEQ * DMODEL + h * HDIM;  // row stride DMODEL

  const int tid = threadIdx.x;
  const int lane = tid & 31;
  const int wid = tid >> 5;
  const int tm = wid >> 1;            // row tile 0..3
  const int tn0 = (wid & 1) * 2;      // col tiles {tn0, tn0+1}

  for (int i = tid; i < HDIM * KVS; i += 256) KVt[i] = 0.f;
  if (tid < HDIM) kcum[tid] = 0.f;
  __syncthreads();

  for (int c = 0; c < SEQ / CHUNK; ++c) {
    const int s0 = c * CHUNK;
    // ---- load chunk: Qs, Ks (+transpose Kts), Vts (transposed) ----
    {
      const int flat = tid * 16;            // 4096 elems / 256 threads
      const int row = flat >> 6, col = flat & 63;
      const v8bf* q0 = (const v8bf*)(Qg + (size_t)s0 * HDIM + flat);
      v8bf qa = q0[0], qb = q0[1];
      *(v8bf*)(Qs + row * STR + col) = qa;
      *(v8bf*)(Qs + row * STR + col + 8) = qb;
      const v8bf* k0 = (const v8bf*)(Kg + (size_t)s0 * HDIM + flat);
      v8bf ka = k0[0], kb = k0[1];
      *(v8bf*)(Ks + row * STR + col) = ka;
      *(v8bf*)(Ks + row * STR + col + 8) = kb;
#pragma unroll
      for (int i = 0; i < 8; ++i) {
        Kts[(col + i) * STR + row] = ka[i];
        Kts[(col + 8 + i) * STR + row] = kb[i];
      }
      const v8bf* v0 = (const v8bf*)(Vg + (size_t)s0 * HDIM + flat);
      v8bf va = v0[0], vb = v0[1];
#pragma unroll
      for (int i = 0; i < 8; ++i) {
        Vts[(col + i) * STR + row] = va[i];
        Vts[(col + 8 + i) * STR + row] = vb[i];
      }
    }
    __syncthreads();

    // ---- S = Q K^T, causal mask within chunk, store bf16 ----
#pragma unroll
    for (int j = 0; j < 2; ++j) {
      const int tn = tn0 + j;
      v8f sacc = {};
#pragma unroll
      for (int kk = 0; kk < HDIM; kk += 32) {
        v16bf a = ldsA(Qs, STR, tm * 16 + (lane & 15), kk, lane);
        v16bf bb = ldsB(Ks, STR, tn * 16 + (lane & 15), kk, lane); // Ks is BT of K^T
        sacc = wmma_bf16(a, bb, sacc);
      }
      const int s = tn * 16 + (lane & 15);
#pragma unroll
      for (int r = 0; r < 8; ++r) {
        const int t = tm * 16 + r + 8 * (lane >> 4);
        float sv = (s <= t) ? sacc[r] : 0.f;
        Ss[t * STR + s] = (__bf16)sv;
      }
    }
    __syncthreads();

    // ---- den[t] = q_t . k_cum(old) + rowsum(S[t,:])  (fp32) ----
    if (tid < CHUNK) {
      float d = 0.f;
      for (int i = 0; i < HDIM; ++i) d += (float)Qs[tid * STR + i] * kcum[i];
      for (int i = 0; i < CHUNK; ++i) d += (float)Ss[tid * STR + i];
      den[tid] = d;
    }

    // ---- num = Q @ KV(old) + S @ V ----
    v8f nacc[2] = {};
#pragma unroll
    for (int j = 0; j < 2; ++j) {
      const int tn = tn0 + j;
#pragma unroll
      for (int kk = 0; kk < HDIM; kk += 32) {
        v16bf a = ldsA(Qs, STR, tm * 16 + (lane & 15), kk, lane);
        v16bf bb = ldsB_f32(KVt, KVS, tn * 16 + (lane & 15), kk, lane);
        nacc[j] = wmma_bf16(a, bb, nacc[j]);
      }
#pragma unroll
      for (int kk = 0; kk < CHUNK; kk += 32) {
        v16bf a = ldsA(Ss, STR, tm * 16 + (lane & 15), kk, lane);
        v16bf bb = ldsB(Vts, STR, tn * 16 + (lane & 15), kk, lane);
        nacc[j] = wmma_bf16(a, bb, nacc[j]);
      }
    }
    __syncthreads();  // all reads of KVt/kcum done; den visible

    // ---- state update: KVt[e][d] += (K^T V)[d][e]; kcum += colsum(K) ----
#pragma unroll
    for (int j = 0; j < 2; ++j) {
      const int tn = tn0 + j;  // e tile
      v8f uacc = {};
#pragma unroll
      for (int kk = 0; kk < CHUNK; kk += 32) {
        v16bf a = ldsA(Kts, STR, tm * 16 + (lane & 15), kk, lane);  // A[d][s]
        v16bf bb = ldsB(Vts, STR, tn * 16 + (lane & 15), kk, lane); // B col e
        uacc = wmma_bf16(a, bb, uacc);
      }
      const int e = tn * 16 + (lane & 15);
#pragma unroll
      for (int r = 0; r < 8; ++r) {
        const int d = tm * 16 + r + 8 * (lane >> 4);
        KVt[e * KVS + d] += uacc[r];
      }
    }
    if (tid < HDIM) {
      float kc = kcum[tid];
      for (int i = 0; i < CHUNK; ++i) kc += (float)Kts[tid * STR + i];
      kcum[tid] = kc;
    }

    // ---- output: O[t,e] = num / (den + eps) ----
#pragma unroll
    for (int j = 0; j < 2; ++j) {
      const int e = (tn0 + j) * 16 + (lane & 15);
#pragma unroll
      for (int r = 0; r < 8; ++r) {
        const int t = tm * 16 + r + 8 * (lane >> 4);
        Og[(size_t)(s0 + t) * DMODEL + e] = nacc[j][r] / (den[t] + EPSV);
      }
    }
    __syncthreads();  // protect LDS tiles / state before next chunk
  }
}

// ---------------------------------------------------------------------------
// Launch: q/k/v projections (WMMA, phi fused) -> chunked linear attention
// (WMMA) -> output projection (WMMA).
// Workspace: Q,K,V bf16 (8MB each) + O fp32 (16MB) = 40MB.
// ---------------------------------------------------------------------------
extern "C" void kernel_launch(void* const* d_in, const int* in_sizes, int n_in,
                              void* d_out, int out_size, void* d_ws, size_t ws_size,
                              hipStream_t stream) {
  const float* x  = (const float*)d_in[0];
  const float* Wq = (const float*)d_in[1];
  const float* bq = (const float*)d_in[2];
  const float* Wk = (const float*)d_in[3];
  const float* bk = (const float*)d_in[4];
  const float* Wv = (const float*)d_in[5];
  const float* bv = (const float*)d_in[6];
  const float* Wo = (const float*)d_in[7];
  const float* bo = (const float*)d_in[8];

  const size_t elems = (size_t)BATCH * SEQ * DMODEL;      // 4M
  char* ws = (char*)d_ws;
  void* Qb = ws;                                          // bf16, 8MB
  void* Kb = ws + elems * 2;                              // bf16, 8MB
  void* Vb = ws + elems * 4;                              // bf16, 8MB
  float* Ob = (float*)(ws + elems * 6);                   // fp32, 16MB

  const int M = BATCH * SEQ;   // 4096
  const int N = DMODEL;        // 1024
  const int K = DMODEL;        // 1024
  dim3 gg(N / 64, M / 128);    // (16, 32)
  dim3 gb(256);

  gemm_bf16_kernel<1><<<gg, gb, 0, stream>>>(x, Wq, bq, Qb, M, N, K);
  gemm_bf16_kernel<1><<<gg, gb, 0, stream>>>(x, Wk, bk, Kb, M, N, K);
  gemm_bf16_kernel<2><<<gg, gb, 0, stream>>>(x, Wv, bv, Vb, M, N, K);

  linattn_kernel<<<dim3(BATCH * NHEAD), dim3(256), 0, stream>>>(
      (const __bf16*)Qb, (const __bf16*)Kb, (const __bf16*)Vb, Ob);

  gemm_bf16_kernel<0><<<gg, gb, 0, stream>>>(Ob, Wo, bo, d_out, M, N, K);
}
